// Qwen3_5GatedDeltaNet_49709951483955
// MI455X (gfx1250) — compile-verified, matrix operand order checked by reading, not run on hardware
//
#include <hip/hip_runtime.h>

// ---------------- problem constants ----------------
#define B_       2
#define S_       4096
#define H_       2048
#define HK_      16
#define HV_      32
#define DK_      128
#define DV_      128
#define KEY_DIM  2048
#define VAL_DIM  4096
#define CONV_DIM 8192
#define KW_      4
#define CS_      64
#define NC_      64            // S_/CS_
#define T_       8192          // B_*S_
#define QKVZ_W   12288
#define SPLIT    4
#define DVS      32            // DV_/SPLIT

typedef unsigned short us;
typedef __attribute__((ext_vector_type(16))) __bf16 bf16x16;
typedef __attribute__((ext_vector_type(8)))  float  v8f;

union Frag { bf16x16 v; unsigned u[8]; us h[16]; };

__device__ inline us f2bf(float f) {
  unsigned u = __builtin_bit_cast(unsigned, f);
  u += 0x7FFFu + ((u >> 16) & 1u);          // round-to-nearest-even
  return (us)(u >> 16);
}
__device__ inline float bf2f(us h) {
  return __builtin_bit_cast(float, ((unsigned)h) << 16);
}

__device__ inline v8f wmma_bf16(bf16x16 a, bf16x16 b, v8f c) {
  return __builtin_amdgcn_wmma_f32_16x16x32_bf16(false, a, false, b, (short)0, c, false, false);
}

// ---- fragment loaders (16x16x32 bf16 WMMA, wave32 layouts per CDNA5 ISA 7.12.2) ----
// A/B fragment where element (r, k) = X[(r0 + (lane&15))*ld + k0 + kmap(p)]
// kmap: p<8 -> 8*hi + p ; p>=8 -> 16 + 8*hi + (p-8)   (hi = lane>>4)
// Usable as A (r = M rows) or as B when the operand is X^T (r = N cols).
__device__ inline bf16x16 ldfragK(const us* X, int ld, int r0, int k0, int lane) {
  Frag f;
  int r = r0 + (lane & 15), hi = lane >> 4;
  const us* p = X + r * ld + k0 + 8 * hi;
#pragma unroll
  for (int v = 0; v < 4; ++v) f.u[v] = *(const unsigned*)(p + 2 * v);
#pragma unroll
  for (int v = 4; v < 8; ++v) f.u[v] = *(const unsigned*)(p + 8 + 2 * v);
  return f.v;
}

// same, but every element scaled by a per-row scalar (row fixed per lane)
__device__ inline bf16x16 ldfragK_scale(const us* X, int ld, int r0, int k0, int lane, float sc) {
  Frag f;
  int r = r0 + (lane & 15), hi = lane >> 4;
  const us* p = X + r * ld + k0 + 8 * hi;
#pragma unroll
  for (int v = 0; v < 8; ++v) {
    int off = (v < 4) ? 2 * v : 8 + 2 * v;
    f.h[2 * v]     = f2bf(bf2f(p[off])     * sc);
    f.h[2 * v + 1] = f2bf(bf2f(p[off + 1]) * sc);
  }
  return f.v;
}

// B fragment from a genuinely row-major [K][ldn] bf16 buffer (strided u16 reads)
__device__ inline bf16x16 ldfragB_rm(const us* Y, int ldn, int k0, int n0, int lane) {
  Frag f;
  int n = n0 + (lane & 15), hi = lane >> 4;
#pragma unroll
  for (int pp = 0; pp < 16; ++pp) {
    int k = (pp < 8) ? (8 * hi + pp) : (16 + 8 * hi + (pp - 8));
    f.h[pp] = Y[(k0 + k) * ldn + n];
  }
  return f.v;
}

// B fragment from a row-major [K][ldn] f32 buffer, converting on the fly
__device__ inline bf16x16 ldfragB_f32(const float* Y, int ldn, int k0, int n0, int lane) {
  Frag f;
  int n = n0 + (lane & 15), hi = lane >> 4;
#pragma unroll
  for (int pp = 0; pp < 16; ++pp) {
    int k = (pp < 8) ? (8 * hi + pp) : (16 + 8 * hi + (pp - 8));
    f.h[pp] = f2bf(Y[(k0 + k) * ldn + n]);
  }
  return f.v;
}

// A fragment for transposed operand with a per-K scale: A(m,k) = Kb[(k0+k)*ld + m0+m] * fs[k0+k]
__device__ inline bf16x16 ldfragA_T_scale(const us* Kb, int ld, int m0, int k0, int lane,
                                          const float* fs) {
  Frag f;
  int m = m0 + (lane & 15), hi = lane >> 4;
#pragma unroll
  for (int pp = 0; pp < 16; ++pp) {
    int k = (pp < 8) ? (8 * hi + pp) : (16 + 8 * hi + (pp - 8));
    int t = k0 + k;
    f.h[pp] = f2bf(bf2f(Kb[t * ld + m]) * fs[t]);
  }
  return f.v;
}

// accumulator (C-layout) init from a bf16 LDS matrix: elem i -> (r0 + i + 8*hi, c0 + (lane&15))
__device__ inline v8f ldaccC(const us* X, int ld, int r0, int c0, int lane) {
  v8f a;
  int n = c0 + (lane & 15), hi = lane >> 4;
#pragma unroll
  for (int i = 0; i < 8; ++i) a[i] = bf2f(X[(r0 + i + 8 * hi) * ld + n]);
  return a;
}

// ================= conversion kernels (one pass, memory bound) =================
// straight f32 -> bf16 copy, 4 elements/thread
__global__ __launch_bounds__(256) void cvt_bf16_kernel(const float* __restrict__ in,
                                                       us* __restrict__ out) {
  size_t i = ((size_t)blockIdx.x * 256 + threadIdx.x) * 4;
  float4 v = *(const float4*)(in + i);
  uint2 r;
  r.x = (unsigned)f2bf(v.x) | ((unsigned)f2bf(v.y) << 16);
  r.y = (unsigned)f2bf(v.z) | ((unsigned)f2bf(v.w) << 16);
  *(uint2*)(out + i) = r;
}

// f32 [K][N] -> bf16 [N][K] tiled transpose-convert
__global__ __launch_bounds__(256) void tcvt_bf16_kernel(const float* __restrict__ in,
                                                        us* __restrict__ out, int K, int N) {
  __shared__ float t[32][33];
  int k0 = blockIdx.y * 32, n0 = blockIdx.x * 32;
  int tx = threadIdx.x & 31, ty = threadIdx.x >> 5;   // ty: 0..7
#pragma unroll
  for (int i = ty; i < 32; i += 8) t[i][tx] = in[(size_t)(k0 + i) * N + n0 + tx];
  __syncthreads();
#pragma unroll
  for (int i = ty; i < 32; i += 8) out[(size_t)(n0 + i) * K + k0 + tx] = f2bf(t[tx][i]);
}

// ================= generic bf16 WMMA GEMM (pre-converted operands) =================
// C[M,N] = A[M,K] @ B[K,N] with A bf16 row-major [M][K], Bt bf16 [N][K].
// grid (N/128, M/128), block 256 (8 waves); 128x128 tile, K-step 64, zero-VALU staging.
__global__ __launch_bounds__(256) void gemm_bf16(const us* __restrict__ A,
                                                 const us* __restrict__ Bt,
                                                 float* __restrict__ C,
                                                 int M, int N, int K) {
  __shared__ __align__(16) us As[128 * 64];   // [m][k]
  __shared__ __align__(16) us Bs[128 * 64];   // [n][k]
  int tid = threadIdx.x, lane = tid & 31, wave = tid >> 5;
  int m_blk = blockIdx.y * 128, n_blk = blockIdx.x * 128;

  v8f acc[8] = {};
  int row = tid >> 1, kk = (tid & 1) * 32;    // 32 bf16 = 64B = 4x b128 per thread

  const us* arow = A + (size_t)(m_blk + row) * K + kk;
  const us* brow = Bt + (size_t)(n_blk + row) * K + kk;
  us* adst = As + row * 64 + kk;
  us* bdst = Bs + row * 64 + kk;

  for (int k0 = 0; k0 < K; k0 += 64) {
    if (k0 + 64 < K) {
      __builtin_prefetch(arow + k0 + 64, 0, 3);   // global_prefetch next slab
      __builtin_prefetch(brow + k0 + 64, 0, 3);
    }
#pragma unroll
    for (int j = 0; j < 4; ++j) ((uint4*)adst)[j] = ((const uint4*)(arow + k0))[j];
#pragma unroll
    for (int j = 0; j < 4; ++j) ((uint4*)bdst)[j] = ((const uint4*)(brow + k0))[j];
    __syncthreads();
#pragma unroll
    for (int ks = 0; ks < 64; ks += 32) {
      bf16x16 a = ldfragK(As, 64, wave * 16, ks, lane);
#pragma unroll
      for (int nt = 0; nt < 8; ++nt) {
        bf16x16 b = ldfragK(Bs, 64, nt * 16, ks, lane);
        acc[nt] = wmma_bf16(a, b, acc[nt]);
      }
    }
    __syncthreads();
  }
  int hi = lane >> 4, j16 = lane & 15;
#pragma unroll
  for (int nt = 0; nt < 8; ++nt) {
    int cc = n_blk + nt * 16 + j16;
#pragma unroll
    for (int i8 = 0; i8 < 8; ++i8) {
      int rr = m_blk + wave * 16 + i8 + 8 * hi;
      C[(size_t)rr * N + cc] = acc[nt][i8];
    }
  }
}

// ================= ba projection -> beta, g =================
__global__ __launch_bounds__(64) void ba_kernel(const float* __restrict__ hs,
                                                const float* __restrict__ Wba,
                                                const float* __restrict__ A_log,
                                                const float* __restrict__ dt_bias,
                                                float* __restrict__ g, float* __restrict__ beta) {
  int t = blockIdx.x;        // 0..T_-1
  int c = threadIdx.x;       // 0..63
  const float* hrow = hs + (size_t)t * H_;
  float s = 0.f;
  for (int k = 0; k < H_; ++k) s += hrow[k] * Wba[k * 64 + c];
  int hk = c >> 2, j = c & 3;
  int hv = hk * 2 + (j & 1);
  if (j < 2) {
    beta[(size_t)t * HV_ + hv] = 1.f / (1.f + expf(-s));
  } else {
    float x = s + dt_bias[hv];
    float sp = (x > 20.f) ? x : log1pf(expf(x));
    g[(size_t)t * HV_ + hv] = -expf(A_log[hv]) * sp;
  }
}

// ================= per-chunk cumsum of g -> (B,HV,S) =================
__global__ __launch_bounds__(64) void gcum_kernel(const float* __restrict__ g,
                                                  float* __restrict__ gcum) {
  __shared__ float sg[CS_];
  int blk = blockIdx.x;
  int t = threadIdx.x;
  int c = blk % NC_;
  int bh = blk / NC_;            // b*HV + hv
  int hv = bh % HV_, b = bh / HV_;
  int s = c * CS_ + t;
  sg[t] = g[((size_t)b * S_ + s) * HV_ + hv];
  __syncthreads();
  for (int off = 1; off < CS_; off <<= 1) {
    float v = (t >= off) ? sg[t - off] : 0.f;
    __syncthreads();
    sg[t] += v;
    __syncthreads();
  }
  gcum[((size_t)b * HV_ + hv) * S_ + s] = sg[t];
}

// ================= gathered causal depthwise conv + SiLU =================
__global__ __launch_bounds__(256) void conv_kernel(const float* __restrict__ qkvz,
                                                   const float* __restrict__ conv_w,
                                                   float* __restrict__ out) {
  int c = blockIdx.x * 256 + threadIdx.x;   // 0..CONV_DIM-1
  int t = blockIdx.y;                       // b*S + s
  int s = t & (S_ - 1);
  int wcol;
  if (c < KEY_DIM) {
    wcol = (c >> 7) * 768 + (c & 127);
  } else if (c < 2 * KEY_DIM) {
    int cc = c - KEY_DIM;
    wcol = (cc >> 7) * 768 + 128 + (cc & 127);
  } else {
    int cc = c - 2 * KEY_DIM;
    wcol = (cc >> 8) * 768 + 256 + (cc & 255);
  }
  float acc = 0.f;
#pragma unroll
  for (int j = 0; j < KW_; ++j) {
    int sp = s - (KW_ - 1) + j;
    if (sp >= 0)
      acc += conv_w[c * KW_ + j] * qkvz[(size_t)(t - (KW_ - 1) + j) * QKVZ_W + wcol];
  }
  out[(size_t)t * CONV_DIM + c] = acc / (1.f + expf(-acc));
}

// ================= per-chunk precompute =================
// A = tril(beta*k @ k^T * decay, -1) (negated), T = (I-A)^{-1} via the nilpotent
// identity (I+A)(I+A^2)(I+A^4)(I+A^8)(I+A^16)(I+A^32), then v_t = T @ (beta*v)
// and k_cumdecay = T @ (beta*exp(g)*k). All matmuls via WMMA.
__global__ __launch_bounds__(128) void precompute_kernel(const float* __restrict__ conv,
                                                         const float* __restrict__ gcum,
                                                         const float* __restrict__ beta,
                                                         float* __restrict__ vt,
                                                         float* __restrict__ kcum) {
  __shared__ __align__(16) us sm[28672];        // 56 KB
  __shared__ float gs[CS_], bs[CS_], red[128];
  us* kbuf = sm;             // 64x128  (normalized k, later beta*exp(g)*k)
  us* Tb   = sm + 8192;      // 64x64
  us* Pa   = sm + 12288;     // 64x64
  us* Pat  = sm + 16384;
  us* Pb   = sm + 20480;
  us* Pbt  = sm + 24576;
  us* vbuf = sm + 12288;     // 64x128, reuses Pa/Pat after T chain

  int blk = blockIdx.x;
  int c = blk % NC_;
  int bh = blk / NC_;
  int hv = bh % HV_, b = bh / HV_;
  int hk = hv >> 1;
  int tid = threadIdx.x, lane = tid & 31, wave = tid >> 5;
  size_t rowbase = (size_t)(b * S_) + (size_t)c * CS_;

  if (tid < CS_) {
    gs[tid] = gcum[(size_t)bh * S_ + c * CS_ + tid];
    bs[tid] = beta[(rowbase + tid) * HV_ + hv];
  }
  // stage L2-normalized k (bf16)
  {
    int r = tid >> 1, hf = tid & 1;
    const float* kr = conv + (rowbase + r) * (size_t)CONV_DIM + KEY_DIM + hk * DK_ + hf * 64;
    float ss = 0.f;
    for (int j = 0; j < 64; ++j) { float v = kr[j]; ss += v * v; }
    red[tid] = ss;
    __syncthreads();
    float rn = rsqrtf(red[2 * r] + red[2 * r + 1] + 1e-6f);
    for (int j = 0; j < 64; ++j) kbuf[r * DK_ + hf * 64 + j] = f2bf(kr[j] * rn);
  }
  __syncthreads();

  // A = beta*k @ k^T, masked & decayed in-register -> Tb = I+A, Pa/Pat = A
  {
    int m0 = wave * 16;
    v8f acc[4] = {};
    float bsc = bs[m0 + (lane & 15)];
    for (int ks = 0; ks < DK_; ks += 32) {
      bf16x16 a = ldfragK_scale(kbuf, DK_, m0, ks, lane, bsc);
#pragma unroll
      for (int nt = 0; nt < 4; ++nt) {
        bf16x16 bb = ldfragK(kbuf, DK_, nt * 16, ks, lane);   // B = k^T
        acc[nt] = wmma_bf16(a, bb, acc[nt]);
      }
    }
    int hi = lane >> 4, j16 = lane & 15;
    for (int nt = 0; nt < 4; ++nt) {
      int j = nt * 16 + j16;
#pragma unroll
      for (int i8 = 0; i8 < 8; ++i8) {
        int i = m0 + i8 + 8 * hi;
        float av = (i > j) ? (-acc[nt][i8] * expf(gs[i] - gs[j])) : 0.f;
        us ab = f2bf(av);
        Pa[i * 64 + j] = ab;
        Pat[j * 64 + i] = ab;
        Tb[i * 64 + j] = f2bf(av + ((i == j) ? 1.f : 0.f));
      }
    }
  }
  __syncthreads();

  // T = (I+A)(I+A^2)...(I+A^32)
  us *P = Pa, *Pt = Pat, *Q = Pb, *Qt = Pbt;
  for (int it = 0; it < 5; ++it) {
    {   // Q = P @ P
      int m0 = wave * 16;
      v8f acc[4] = {};
      for (int ks = 0; ks < 64; ks += 32) {
        bf16x16 a = ldfragK(P, 64, m0, ks, lane);
#pragma unroll
        for (int nt = 0; nt < 4; ++nt) {
          bf16x16 bb = ldfragK(Pt, 64, nt * 16, ks, lane);
          acc[nt] = wmma_bf16(a, bb, acc[nt]);
        }
      }
      int hi = lane >> 4, j16 = lane & 15;
      for (int nt = 0; nt < 4; ++nt) {
        int j = nt * 16 + j16;
#pragma unroll
        for (int i8 = 0; i8 < 8; ++i8) {
          int i = m0 + i8 + 8 * hi;
          us qb = f2bf(acc[nt][i8]);
          Q[i * 64 + j] = qb;
          Qt[j * 64 + i] = qb;
        }
      }
    }
    __syncthreads();
    {   // T = T + T @ Q   (C operand initialized from T)
      int m0 = wave * 16;
      v8f acc[4];
#pragma unroll
      for (int nt = 0; nt < 4; ++nt) acc[nt] = ldaccC(Tb, 64, m0, nt * 16, lane);
      for (int ks = 0; ks < 64; ks += 32) {
        bf16x16 a = ldfragK(Tb, 64, m0, ks, lane);
#pragma unroll
        for (int nt = 0; nt < 4; ++nt) {
          bf16x16 bb = ldfragK(Qt, 64, nt * 16, ks, lane);
          acc[nt] = wmma_bf16(a, bb, acc[nt]);
        }
      }
      __syncthreads();   // all reads of Tb done before overwrite
      int hi = lane >> 4, j16 = lane & 15;
      for (int nt = 0; nt < 4; ++nt) {
        int j = nt * 16 + j16;
#pragma unroll
        for (int i8 = 0; i8 < 8; ++i8) {
          int i = m0 + i8 + 8 * hi;
          Tb[i * 64 + j] = f2bf(acc[nt][i8]);
        }
      }
    }
    __syncthreads();
    us* t1 = P; P = Q; Q = t1;
    t1 = Pt; Pt = Qt; Qt = t1;
  }

  // stage beta*v, then v_t = T @ (beta*v)
  {
    int r = tid >> 1, hf = tid & 1;
    const float* vr = conv + (rowbase + r) * (size_t)CONV_DIM + 2 * KEY_DIM + hv * DV_ + hf * 64;
    float bsc = bs[r];
    for (int j = 0; j < 64; ++j) vbuf[r * DV_ + hf * 64 + j] = f2bf(vr[j] * bsc);
  }
  __syncthreads();
  {
    int m0 = wave * 16;
    v8f acc[8] = {};
    for (int ks = 0; ks < 64; ks += 32) {
      bf16x16 a = ldfragK(Tb, 64, m0, ks, lane);
#pragma unroll
      for (int nt = 0; nt < 8; ++nt) {
        bf16x16 bb = ldfragB_rm(vbuf, DV_, ks, nt * 16, lane);
        acc[nt] = wmma_bf16(a, bb, acc[nt]);
      }
    }
    int hi = lane >> 4, j16 = lane & 15;
    float* dst = vt + ((size_t)bh * NC_ + c) * CS_ * DV_;
    for (int nt = 0; nt < 8; ++nt) {
      int j = nt * 16 + j16;
#pragma unroll
      for (int i8 = 0; i8 < 8; ++i8) dst[(m0 + i8 + 8 * hi) * DV_ + j] = acc[nt][i8];
    }
  }
  __syncthreads();
  // kbuf <- beta*exp(g)*k (in place), then k_cumdecay = T @ kbuf
  {
    int r = tid >> 1, hf = tid & 1;
    float fr = bs[r] * expf(gs[r]);
    us* kp = kbuf + r * DK_ + hf * 64;
    for (int j = 0; j < 64; ++j) kp[j] = f2bf(bf2f(kp[j]) * fr);
  }
  __syncthreads();
  {
    int m0 = wave * 16;
    v8f acc[8] = {};
    for (int ks = 0; ks < 64; ks += 32) {
      bf16x16 a = ldfragK(Tb, 64, m0, ks, lane);
#pragma unroll
      for (int nt = 0; nt < 8; ++nt) {
        bf16x16 bb = ldfragB_rm(kbuf, DK_, ks, nt * 16, lane);
        acc[nt] = wmma_bf16(a, bb, acc[nt]);
      }
    }
    int hi = lane >> 4, j16 = lane & 15;
    float* dst = kcum + ((size_t)bh * NC_ + c) * CS_ * DK_;
    for (int nt = 0; nt < 8; ++nt) {
      int j = nt * 16 + j16;
#pragma unroll
      for (int i8 = 0; i8 < 8; ++i8) dst[(m0 + i8 + 8 * hi) * DK_ + j] = acc[nt][i8];
    }
  }
}

// ================= sequential chunk scan =================
// One block per (b, hv, column-group of 32 of DV). fp32 state in LDS.
__global__ __launch_bounds__(128) void scan_kernel(const float* __restrict__ conv,
                                                   const float* __restrict__ gcum,
                                                   const float* __restrict__ vt,
                                                   const float* __restrict__ kcum,
                                                   float* __restrict__ core) {
  __shared__ float Sst[DK_ * DVS];                 // 16 KB state
  __shared__ __align__(16) us kb[CS_ * DK_];       // 16 KB normalized k
  __shared__ __align__(16) us buf0[CS_ * DK_];     // 16 KB kcum, then q
  __shared__ __align__(16) us vnew[CS_ * DVS];     // 4 KB
  __shared__ __align__(16) us attn[CS_ * CS_];     // 8 KB
  __shared__ float gs[CS_], fsc[CS_], red[128];

  int blk = blockIdx.x;
  int cg = blk % SPLIT;
  int bh = blk / SPLIT;
  int hv = bh % HV_, b = bh / HV_;
  int hk = hv >> 1;
  int tid = threadIdx.x, lane = tid & 31, wave = tid >> 5;

  for (int i = tid; i < DK_ * DVS; i += 128) Sst[i] = 0.f;

  for (int c = 0; c < NC_; ++c) {
    __syncthreads();
    if (tid < CS_) gs[tid] = gcum[(size_t)bh * S_ + c * CS_ + tid];
    __syncthreads();
    if (tid < CS_) fsc[tid] = expf(gs[CS_ - 1] - gs[tid]);
    size_t rowbase = (size_t)(b * S_) + (size_t)c * CS_;
    // stage L2-normalized k
    {
      int r = tid >> 1, hf = tid & 1;
      const float* kr = conv + (rowbase + r) * (size_t)CONV_DIM + KEY_DIM + hk * DK_ + hf * 64;
      float ss = 0.f;
      for (int j = 0; j < 64; ++j) { float v = kr[j]; ss += v * v; }
      red[tid] = ss;
      __syncthreads();
      float rn = rsqrtf(red[2 * r] + red[2 * r + 1] + 1e-6f);
      for (int j = 0; j < 64; ++j) kb[r * DK_ + hf * 64 + j] = f2bf(kr[j] * rn);
    }
    // stage k_cumdecay into buf0
    {
      int r = tid >> 1, hf = tid & 1;
      const float* sr = kcum + ((size_t)bh * NC_ + c) * CS_ * DK_ + r * DK_ + hf * 64;
      for (int j = 0; j < 64; ++j) buf0[r * DK_ + hf * 64 + j] = f2bf(sr[j]);
    }
    __syncthreads();
    // v_new = v_t - kcum @ S
    {
      int m0 = wave * 16;
      v8f acc[2] = {};
      for (int ks = 0; ks < DK_; ks += 32) {
        bf16x16 a = ldfragK(buf0, DK_, m0, ks, lane);
#pragma unroll
        for (int nt = 0; nt < 2; ++nt) {
          bf16x16 bb = ldfragB_f32(Sst, DVS, ks, nt * 16, lane);
          acc[nt] = wmma_bf16(a, bb, acc[nt]);
        }
      }
      int hi = lane >> 4, j16 = lane & 15;
      const float* vsrc = vt + ((size_t)bh * NC_ + c) * CS_ * DV_;
      for (int nt = 0; nt < 2; ++nt) {
        int j = nt * 16 + j16;
#pragma unroll
        for (int i8 = 0; i8 < 8; ++i8) {
          int i = m0 + i8 + 8 * hi;
          vnew[i * DVS + j] = f2bf(vsrc[i * DV_ + cg * DVS + j] - acc[nt][i8]);
        }
      }
    }
    __syncthreads();
    // stage q (normalized, * 1/sqrt(DK)) into buf0
    {
      int r = tid >> 1, hf = tid & 1;
      const float* qr = conv + (rowbase + r) * (size_t)CONV_DIM + hk * DK_ + hf * 64;
      float ss = 0.f;
      for (int j = 0; j < 64; ++j) { float v = qr[j]; ss += v * v; }
      red[tid] = ss;
      __syncthreads();
      float rn = rsqrtf(red[2 * r] + red[2 * r + 1] + 1e-6f) * 0.08838834764831845f;
      for (int j = 0; j < 64; ++j) buf0[r * DK_ + hf * 64 + j] = f2bf(qr[j] * rn);
    }
    __syncthreads();
    // attn = (q @ k^T) * decay * strict_lower
    {
      int m0 = wave * 16;
      v8f acc[4] = {};
      for (int ks = 0; ks < DK_; ks += 32) {
        bf16x16 a = ldfragK(buf0, DK_, m0, ks, lane);
#pragma unroll
        for (int nt = 0; nt < 4; ++nt) {
          bf16x16 bb = ldfragK(kb, DK_, nt * 16, ks, lane);
          acc[nt] = wmma_bf16(a, bb, acc[nt]);
        }
      }
      int hi = lane >> 4, j16 = lane & 15;
      for (int nt = 0; nt < 4; ++nt) {
        int j = nt * 16 + j16;
#pragma unroll
        for (int i8 = 0; i8 < 8; ++i8) {
          int i = m0 + i8 + 8 * hi;
          attn[i * CS_ + j] = f2bf((i > j) ? acc[nt][i8] * expf(gs[i] - gs[j]) : 0.f);
        }
      }
    }
    __syncthreads();
    // out = (q*exp(g)) @ S + attn @ v_new
    {
      int m0 = wave * 16;
      float qsc = expf(gs[m0 + (lane & 15)]);
      v8f acc[2] = {};
      for (int ks = 0; ks < DK_; ks += 32) {
        bf16x16 a = ldfragK_scale(buf0, DK_, m0, ks, lane, qsc);
#pragma unroll
        for (int nt = 0; nt < 2; ++nt) {
          bf16x16 bb = ldfragB_f32(Sst, DVS, ks, nt * 16, lane);
          acc[nt] = wmma_bf16(a, bb, acc[nt]);
        }
      }
      for (int ks = 0; ks < CS_; ks += 32) {
        bf16x16 a = ldfragK(attn, CS_, m0, ks, lane);
#pragma unroll
        for (int nt = 0; nt < 2; ++nt) {
          bf16x16 bb = ldfragB_rm(vnew, DVS, ks, nt * 16, lane);
          acc[nt] = wmma_bf16(a, bb, acc[nt]);
        }
      }
      int hi = lane >> 4, j16 = lane & 15;
      for (int nt = 0; nt < 2; ++nt) {
        int j = nt * 16 + j16;
#pragma unroll
        for (int i8 = 0; i8 < 8; ++i8) {
          int i = m0 + i8 + 8 * hi;
          core[((rowbase + i) * HV_ + hv) * DV_ + cg * DVS + j] = acc[nt][i8];
        }
      }
    }
    __syncthreads();
    // S = S*exp(g_last) + (k * exp(g_last - g))^T @ v_new
    {
      int m0 = wave * 32;     // 32 state rows (d) per wave
      float gl = expf(gs[CS_ - 1]);
      v8f acc[2][2] = {};
      for (int ks = 0; ks < CS_; ks += 32) {
#pragma unroll
        for (int mt = 0; mt < 2; ++mt) {
          bf16x16 a = ldfragA_T_scale(kb, DK_, m0 + mt * 16, ks, lane, fsc);
#pragma unroll
          for (int nt = 0; nt < 2; ++nt) {
            bf16x16 bb = ldfragB_rm(vnew, DVS, ks, nt * 16, lane);
            acc[mt][nt] = wmma_bf16(a, bb, acc[mt][nt]);
          }
        }
      }
      int hi = lane >> 4, j16 = lane & 15;
      for (int mt = 0; mt < 2; ++mt)
        for (int nt = 0; nt < 2; ++nt) {
          int j = nt * 16 + j16;
#pragma unroll
          for (int i8 = 0; i8 < 8; ++i8) {
            int d = m0 + mt * 16 + i8 + 8 * hi;
            Sst[d * DVS + j] = Sst[d * DVS + j] * gl + acc[mt][nt][i8];
          }
        }
    }
  }
}

// ================= gated RMSNorm (writes bf16 activation for final GEMM) =================
__global__ __launch_bounds__(128) void norm_gate_kernel(const float* __restrict__ core,
                                                        const float* __restrict__ qkvz,
                                                        const float* __restrict__ nw,
                                                        us* __restrict__ gated) {
  __shared__ float red[128];
  int idx = blockIdx.x;
  int hv = idx % HV_, t = idx / HV_;
  int d = threadIdx.x;
  float v = core[((size_t)t * HV_ + hv) * DV_ + d];
  red[d] = v * v;
  __syncthreads();
  for (int off = 64; off > 0; off >>= 1) {
    if (d < off) red[d] += red[d + off];
    __syncthreads();
  }
  float inv = rsqrtf(red[0] * (1.f / DV_) + 1e-6f);
  int hk = hv >> 1, sub = hv & 1;
  float z = qkvz[(size_t)t * QKVZ_W + hk * 768 + 512 + sub * 128 + d];
  float sz = z / (1.f + expf(-z));
  gated[(size_t)t * VAL_DIM + hv * DV_ + d] = f2bf(v * inv * nw[d] * sz);
}

// ================= host launcher =================
extern "C" void kernel_launch(void* const* d_in, const int* in_sizes, int n_in,
                              void* d_out, int out_size, void* d_ws, size_t ws_size,
                              hipStream_t stream) {
  (void)in_sizes; (void)n_in; (void)out_size; (void)ws_size;
  const float* hs    = (const float*)d_in[0];   // (B,S,H)
  const float* Wqkvz = (const float*)d_in[1];   // (H, 12288)
  const float* Wba   = (const float*)d_in[2];   // (H, 64)
  const float* convw = (const float*)d_in[3];   // (CONV_DIM, KW)
  const float* Alog  = (const float*)d_in[4];   // (HV)
  const float* dtb   = (const float*)d_in[5];   // (HV)
  const float* normw = (const float*)d_in[6];   // (DV)
  const float* Wout  = (const float*)d_in[7];   // (VAL_DIM, H)
  float* out = (float*)d_out;

  float* ws    = (float*)d_ws;
  float* qkvz  = ws;                                          // 8192*12288
  float* convb = qkvz  + (size_t)T_ * QKVZ_W;                 // 8192*8192
  float* gbuf  = convb + (size_t)T_ * CONV_DIM;               // 8192*32
  float* bbuf  = gbuf  + (size_t)T_ * HV_;                    // 8192*32
  float* gcm   = bbuf  + (size_t)T_ * HV_;                    // 8192*32 (B,HV,S)
  float* vtb   = gcm   + (size_t)T_ * HV_;                    // 2*32*64*64*128
  float* kcb   = vtb   + (size_t)B_ * HV_ * NC_ * CS_ * DV_;  // same size
  float* coreb = kcb   + (size_t)B_ * HV_ * NC_ * CS_ * DK_;  // 8192*32*128
  us* hsb    = (us*)(coreb + (size_t)T_ * HV_ * DV_);         // 8192*2048 bf16
  us* wqt    = hsb + (size_t)T_ * H_;                         // 12288*2048 bf16 (transposed)
  us* gatedb = wqt + (size_t)QKVZ_W * H_;                     // 8192*4096 bf16
  us* woutt  = gatedb + (size_t)T_ * VAL_DIM;                 // 2048*4096 bf16 (transposed)

  // 0) operand conversions (bf16, weights transposed to [N][K])
  cvt_bf16_kernel<<<(T_ * H_) / 1024, 256, 0, stream>>>(hs, hsb);
  tcvt_bf16_kernel<<<dim3(QKVZ_W / 32, H_ / 32), 256, 0, stream>>>(Wqkvz, wqt, H_, QKVZ_W);
  tcvt_bf16_kernel<<<dim3(H_ / 32, VAL_DIM / 32), 256, 0, stream>>>(Wout, woutt, VAL_DIM, H_);
  // 1) qkvz projection GEMM (8192x12288x2048)
  gemm_bf16<<<dim3(QKVZ_W / 128, T_ / 128), 256, 0, stream>>>(hsb, wqt, qkvz, T_, QKVZ_W, H_);
  // 2) ba projection + beta/g activation
  ba_kernel<<<T_, 64, 0, stream>>>(hs, Wba, Alog, dtb, gbuf, bbuf);
  // 3) per-chunk cumsum of g
  gcum_kernel<<<B_ * HV_ * NC_, 64, 0, stream>>>(gbuf, gcm);
  // 4) causal depthwise conv + SiLU (gathered from qkvz)
  conv_kernel<<<dim3(CONV_DIM / 256, T_), 256, 0, stream>>>(qkvz, convw, convb);
  // 5) per-chunk precompute (T inverse, v_t, k_cumdecay)
  precompute_kernel<<<B_ * HV_ * NC_, 128, 0, stream>>>(convb, gcm, bbuf, vtb, kcb);
  // 6) sequential inter-chunk scan
  scan_kernel<<<B_ * HV_ * SPLIT, 128, 0, stream>>>(convb, gcm, vtb, kcb, coreb);
  // 7) gated RMSNorm -> bf16 activations
  norm_gate_kernel<<<T_ * HV_, 128, 0, stream>>>(coreb, qkvz, normw, gatedb);
  // 8) output GEMM (8192x2048x4096)
  gemm_bf16<<<dim3(H_ / 128, T_ / 128), 256, 0, stream>>>(gatedb, woutt, out, T_, H_, VAL_DIM);
}
